// ManualConv2d_28278064677249
// MI455X (gfx1250) — matrix operand order repaired and checked by reading, not compiled, not run on hardware
//
#include <hip/hip_runtime.h>

// ---------------------------------------------------------------------------
// Conv2d 3x3 s1 p1 : x[16,96,112,112] (*) w[192,96,3,3] + b[192]
// Implicit GEMM on CDNA5 WMMA:  M = 16*112*112 pixels, N = 192, K = 96*9.
// bf16 inputs, f32 accumulate via v_wmma_f32_16x16x32_bf16.
// B (weights) staged per-workgroup through LDS by the Tensor Data Mover
// (tensor_load_to_lds + s_wait_tensorcnt), double-buffered; A prefetched one
// K-step ahead; B fragments depth-3 register-pipelined with
// __builtin_amdgcn_sched_barrier(0) pinning the prefetch above each WMMA so
// waits become partial (s_wait_dscnt 4) instead of wait-0 stalls.
// ---------------------------------------------------------------------------

typedef __attribute__((ext_vector_type(16))) __bf16        v16bf;
typedef __attribute__((ext_vector_type(8)))  float         v8f;
typedef __attribute__((ext_vector_type(4)))  unsigned int  u32x4;
typedef __attribute__((ext_vector_type(8)))  int           i32x8;
typedef __attribute__((ext_vector_type(4)))  int           i32x4;

#define NB   16
#define CI   96
#define HW   112
#define HWP  114          // padded spatial
#define KO   192
#define PIX  (HW * HW)    // 12544 pixels per image
#define KSTEPS 27         // 9 (r,s) positions x 3 channel-chunks of 32
#define BSTEP_BYTES 12288 // 12 B-fragments x 1KB per K-step

__device__ __forceinline__ unsigned short f32_to_bf16(float f) {
    union { float f; unsigned int u; } x; x.f = f;
    unsigned int u = x.u;
    unsigned int r = u + 0x7FFFu + ((u >> 16) & 1u);   // round-to-nearest-even
    return (unsigned short)(r >> 16);
}

// ---------------------------------------------------------------------------
// Kernel 1: NCHW f32 -> zero-padded NHWC bf16  (xpad[n][hp][wp][c])
// ---------------------------------------------------------------------------
__global__ __launch_bounds__(256) void pad_convert_nhwc(
    const float* __restrict__ x, unsigned short* __restrict__ xpad)
{
    const int rowid = blockIdx.x;          // 0 .. NB*HWP-1
    const int n  = rowid / HWP;
    const int hp = rowid % HWP;
    __shared__ float lds[CI * 113];        // 113 stride: odd vs 64 banks
    const bool interior = (hp >= 1 && hp <= HW);

    if (interior) {
        const float* src = x + (size_t)n * CI * PIX + (size_t)(hp - 1) * HW;
        for (int f = threadIdx.x; f < CI * HW; f += blockDim.x) {
            const int c = f / HW, w = f % HW;
            lds[c * 113 + w] = src[(size_t)c * PIX + w];
        }
    }
    __syncthreads();

    unsigned short* dst = xpad + (size_t)rowid * HWP * CI;
    for (int o = threadIdx.x; o < HWP * CI; o += blockDim.x) {
        const int wp = o / CI, c = o % CI;
        float v = 0.0f;
        if (interior && wp >= 1 && wp <= HW) v = lds[c * 113 + (wp - 1)];
        dst[o] = f32_to_bf16(v);
    }
}

// ---------------------------------------------------------------------------
// Kernel 2: pack weights into per-lane WMMA B-fragment order (bf16).
// Fragment id = kstep*12 + ntile ; 32 lanes x 16 bf16 each.
// Dense 32x16 16-bit B layout: lane L holds N = nBase+(L&15),
// element e holds K = (e<8 ? 0 : 16) + (L>=16 ? 8 : 0) + (e&7).
// ---------------------------------------------------------------------------
__global__ __launch_bounds__(256) void pack_weights(
    const float* __restrict__ wgt, unsigned short* __restrict__ bpack)
{
    const int gid  = blockIdx.x * blockDim.x + threadIdx.x;
    const int frag = gid >> 5;
    const int lane = gid & 31;
    if (frag >= KSTEPS * 12) return;
    const int ntile = frag % 12;
    const int kci   = (frag / 12) % 3;
    const int rs    = frag / 36;
    const int r = rs / 3, s = rs % 3;
    const int o   = ntile * 16 + (lane & 15);
    const int lhi = lane >> 4;
    unsigned short* dst = bpack + (size_t)frag * 512 + lane * 16;
    #pragma unroll
    for (int e = 0; e < 16; ++e) {
        const int c = kci * 32 + (e < 8 ? lhi * 8 + e : 16 + lhi * 8 + (e - 8));          // K_gemm channel
        const float v = wgt[(((size_t)o * CI + c) * 3 + r) * 3 + s];
        dst[e] = f32_to_bf16(v);
    }
}

// ---------------------------------------------------------------------------
// TDM: stage one K-step of the B-pack (12KB) into LDS as a 1-D tile.
// D# per CDNA5 ISA ch.8: data_size=8B units, tile_dim0=tensor_dim0=stride=1536.
// lds_addr = addr[31:0] of the generic shared pointer (flat->LDS mapping).
// 6-arg builtin variant (clang-23): (g0, g1, g2, g3, g4, cpol).
// ---------------------------------------------------------------------------
__device__ __forceinline__ void tdm_stage_b(const unsigned short* gsrc,
                                            const unsigned short* ldsdst)
{
    const unsigned long long ga = (unsigned long long)(size_t)gsrc;
    const unsigned lds_addr = (unsigned)(size_t)ldsdst;
    u32x4 g0;
    g0[0] = 1u;                                   // count=1, user load descriptor
    g0[1] = lds_addr;                             // LDS byte address
    g0[2] = (unsigned)(ga & 0xffffffffu);         // global_addr[31:0]
    g0[3] = (unsigned)(ga >> 32) | (2u << 30);    // global_addr[56:32] | type=2
    i32x8 g1;
    g1[0] = (int)(3u << 16);        // workgroup_mask=0, data_size=3 (8B)
    g1[1] = (int)(0x600u << 16);    // tensor_dim0[15:0]=1536 in bits[31:16]
    g1[2] = (int)(1u << 16);        // tensor_dim0[31:16]=0, tensor_dim1=1
    g1[3] = (int)(1536u << 16);     // tile_dim0 = 1536 (x8B = 12288B)
    g1[4] = 1;                      // tile_dim1 = 1, tile_dim2 = 0
    g1[5] = 1536;                   // tensor_dim0_stride[31:0]
    g1[6] = 0;
    g1[7] = 0;
    i32x4 z4 = {0, 0, 0, 0};
    i32x8 z8 = {0, 0, 0, 0, 0, 0, 0, 0};
    __builtin_amdgcn_tensor_load_to_lds(g0, g1, z4, z4, z8, 0);
}

__device__ __forceinline__ int a_koffset(int ks) {
    const int kci = ks % 3;
    const int rs  = ks / 3;
    const int r = rs / 3, s = rs % 3;
    return (r * HWP + s) * CI + kci * 32;   // bf16-element offset into xpad row
}

// ---------------------------------------------------------------------------
// Kernel 3: main implicit-GEMM conv. 8 waves/block; wave = 16 pixels x 192
// out-channels (12 f32 accumulators). 27 K-steps x 12 WMMA each.
// A-fragment (16-bit A 16x32 layout): lane holds row M=lane&15,
//   elements 0..7  -> K = kbase + lhi*8 + 0..7        (one b128 load)
//   elements 8..15 -> K = kbase + 16 + lhi*8 + 0..7   (one b128 load)
// ---------------------------------------------------------------------------
__global__ __launch_bounds__(256) void conv3x3_wmma_bf16(
    const unsigned short* __restrict__ xpad,
    const unsigned short* __restrict__ bpack,
    const float* __restrict__ bias,
    float* __restrict__ out)
{
    __shared__ __align__(16) unsigned short bsmem[2][BSTEP_BYTES / 2]; // 2 x 12KB

    const int lane = threadIdx.x & 31;
    const int wave = threadIdx.x >> 5;
    const int lhi  = lane >> 4;
    const int row  = lane & 15;
    const int pixelBase = (blockIdx.x * 8 + wave) * 16;   // 16-aligned, 112%16==0

    // This lane's A-row pixel -> padded NHWC base address
    const int p   = pixelBase + row;
    const int n   = p / PIX;
    const int rem = p - n * PIX;
    const int h   = rem / HW;
    const int w   = rem - h * HW;
    const size_t aBase = (((size_t)n * HWP + h) * HWP + w) * CI + (size_t)lhi * 8;

    v8f acc[12] = {};

    union AU { uint4 q[2]; v16bf v; };
    AU Acur, Anext;

    // ---- pipeline prologue: TDM-stage B[0], preload A[0] ----
    if (wave == 0) tdm_stage_b(bpack, &bsmem[0][0]);
    {
        const unsigned short* ap = xpad + aBase;          // a_koffset(0) == 0
        Acur.q[0] = *reinterpret_cast<const uint4*>(ap);
        Acur.q[1] = *reinterpret_cast<const uint4*>(ap + 16);
    }
    if (wave == 0) __builtin_amdgcn_s_wait_tensorcnt(0);
    __syncthreads();

    #pragma unroll 1
    for (int ks = 0; ks < KSTEPS; ++ks) {
        const bool more = (ks + 1 < KSTEPS);
        // TDM-stage next K-step's B into the other LDS buffer
        if (wave == 0 && more)
            tdm_stage_b(bpack + (size_t)(ks + 1) * (BSTEP_BYTES / 2),
                        &bsmem[(ks + 1) & 1][0]);
        // Prefetch next K-step's A fragment (global, L2-resident xpad)
        if (more) {
            const unsigned short* ap = xpad + aBase + a_koffset(ks + 1);
            Anext.q[0] = *reinterpret_cast<const uint4*>(ap);
            Anext.q[1] = *reinterpret_cast<const uint4*>(ap + 16);
        }
        // 12 WMMAs; B via depth-3 rotating register buffers. sched_barrier(0)
        // pins each prefetch pair ABOVE the WMMA so 3 buffers stay live and
        // the waitcnt pass emits partial s_wait_dscnt instead of wait-0.
        const unsigned short* bl = &bsmem[ks & 1][0] + lane * 16;
        union BU { uint4 q[2]; v16bf v; } Bf[3];
        Bf[0].q[0] = *reinterpret_cast<const uint4*>(bl);
        Bf[0].q[1] = *reinterpret_cast<const uint4*>(bl + 8);
        Bf[1].q[0] = *reinterpret_cast<const uint4*>(bl + 512);
        Bf[1].q[1] = *reinterpret_cast<const uint4*>(bl + 512 + 8);
        __builtin_amdgcn_sched_barrier(0);
        #pragma unroll
        for (int t = 0; t < 12; ++t) {
            if (t + 2 < 12) {
                const int nxt = (t + 2) % 3;
                Bf[nxt].q[0] = *reinterpret_cast<const uint4*>(bl + (t + 2) * 512);
                Bf[nxt].q[1] = *reinterpret_cast<const uint4*>(bl + (t + 2) * 512 + 8);
            }
            __builtin_amdgcn_sched_barrier(0);
            acc[t] = __builtin_amdgcn_wmma_f32_16x16x32_bf16(
                false, Acur.v, false, Bf[t % 3].v, (short)0, acc[t], false, false);
            __builtin_amdgcn_sched_barrier(0);
        }
        if (wave == 0 && more) __builtin_amdgcn_s_wait_tensorcnt(0);
        __syncthreads();
        if (more) Acur = Anext;
    }

    // Store: lane holds N = row (out channel within tile), M = lhi*8 + g.
    // Pixels lhi*8..+7 are w-contiguous -> two float4 stores per tile.
    const int p0   = pixelBase + lhi * 8;
    const int n2   = p0 / PIX;
    const int rem2 = p0 - n2 * PIX;
    const int h2   = rem2 / HW;
    const int w2   = rem2 - h2 * HW;
    #pragma unroll
    for (int t = 0; t < 12; ++t) {
        const int o = t * 16 + row;
        const float b = bias[o];
        float* op = out + (((size_t)n2 * KO + o) * HW + h2) * HW + w2;
        float4 lo = make_float4(acc[t][0] + b, acc[t][1] + b, acc[t][2] + b, acc[t][3] + b);
        float4 hi = make_float4(acc[t][4] + b, acc[t][5] + b, acc[t][6] + b, acc[t][7] + b);
        *reinterpret_cast<float4*>(op)     = lo;
        *reinterpret_cast<float4*>(op + 4) = hi;
    }
}

// ---------------------------------------------------------------------------
extern "C" void kernel_launch(void* const* d_in, const int* in_sizes, int n_in,
                              void* d_out, int out_size, void* d_ws, size_t ws_size,
                              hipStream_t stream) {
    const float* x    = (const float*)d_in[0];
    const float* wgt  = (const float*)d_in[1];
    const float* bias = (const float*)d_in[2];
    float* out = (float*)d_out;

    // Workspace layout: xpad (NB*HWP*HWP*CI bf16 = ~39.9MB), then bpack (331KB)
    const size_t xpad_elems = (size_t)NB * HWP * HWP * CI;
    unsigned short* xpad  = (unsigned short*)d_ws;
    unsigned short* bpack = xpad + xpad_elems;

    pad_convert_nhwc<<<NB * HWP, 256, 0, stream>>>(x, xpad);
    pack_weights<<<(KSTEPS * 12 * 32 + 255) / 256, 256, 0, stream>>>(wgt, bpack);

    // 12544 pixel-tiles of 16 / 8 waves per block = 1568 blocks (exact cover)
    conv3x3_wmma_bf16<<<1568, 256, 0, stream>>>(xpad, bpack, bias, out);
}
